// TransformerBlock_24051816858268
// MI455X (gfx1250) — matrix operand order repaired and checked by reading, not compiled, not run on hardware
//
#include <hip/hip_runtime.h>
#include <math.h>

typedef __attribute__((ext_vector_type(16))) __bf16 v16bf;
typedef __attribute__((ext_vector_type(8)))  float  v8f;

union BF16x16 {
    v16bf v;
    uint4 q[2];
    unsigned short u[16];
};

__device__ __forceinline__ unsigned short f2bf(float f) {
    unsigned int u = __builtin_bit_cast(unsigned int, f);
    u += 0x7FFFu + ((u >> 16) & 1u);   // round-to-nearest-even
    return (unsigned short)(u >> 16);
}

__device__ __forceinline__ v8f wmma_bf16(v16bf a, v16bf b, v8f c) {
    return __builtin_amdgcn_wmma_f32_16x16x32_bf16(
        /*neg_a=*/false, a, /*neg_b=*/false, b,
        /*c_mod=*/(short)0, c, /*reuse_a=*/false, /*reuse_b=*/false);
}

// B-fragment (32 x 16, 16-bit): lane L = column L%16; elems 0..7 -> K=k0+8*hi+0..7, elems 8..15 -> K=k0+8*hi+16..23
__device__ __forceinline__ v16bf load_bfrag(const unsigned short* base,
                                            int col, int ldk, int k0, int hi) {
    BF16x16 f;
    const unsigned short* p = base + (size_t)col * ldk + k0 + 8 * hi;
    f.q[0] = *(const uint4*)(p);
    f.q[1] = *(const uint4*)(p + 16);
    return f.v;
}

// A-fragment (16 x 32, 16-bit): lane L = row L%16; elems 0..7 -> K=k0+8*hi+0..7,
// elems 8..15 -> K=k0+8*hi+16..23
__device__ __forceinline__ v16bf load_afrag(const unsigned short* base,
                                            int row, int ldk, int k0, int hi) {
    BF16x16 f;
    const unsigned short* p = base + (size_t)row * ldk + k0 + 8 * hi;
    f.q[0] = *(const uint4*)(p);
    f.q[1] = *(const uint4*)(p + 16);
    return f.v;
}

// 16B async copy global -> LDS (ASYNCcnt-tracked, gfx1250 VGLOBAL op 98)
__device__ __forceinline__ void async_cp16(unsigned ldsOff, unsigned long long gaddr) {
    asm volatile("global_load_async_to_lds_b128 %0, %1, off"
                 :: "v"(ldsOff), "v"(gaddr) : "memory");
}

// ---------------------------------------------------------------------------
// Weight cast + transpose: w[K][N] f32 -> wT[N][K] bf16
// ---------------------------------------------------------------------------
__global__ __launch_bounds__(256) void tb_transpose_cast(
    const float* __restrict__ w, unsigned short* __restrict__ wT, int K, int N) {
    long idx = (long)blockIdx.x * 256 + threadIdx.x;
    if (idx < (long)K * N) {
        int k = (int)(idx / N);
        int n = (int)(idx - (long)k * N);
        wT[(size_t)n * K + k] = f2bf(w[idx]);
    }
}

// ---------------------------------------------------------------------------
// LayerNorm (d = 768) f32 -> bf16
// ---------------------------------------------------------------------------
__global__ __launch_bounds__(256) void tb_layernorm(
    const float* __restrict__ x, const float* __restrict__ g,
    const float* __restrict__ b, unsigned short* __restrict__ out) {
    __shared__ float sS[256], sQ[256];
    const int row = blockIdx.x, tid = threadIdx.x;
    const float* xr = x + (size_t)row * 768;
    float v0 = xr[tid], v1 = xr[tid + 256], v2 = xr[tid + 512];
    sS[tid] = v0 + v1 + v2;
    sQ[tid] = v0 * v0 + v1 * v1 + v2 * v2;
    __syncthreads();
    for (int st = 128; st > 0; st >>= 1) {
        if (tid < st) { sS[tid] += sS[tid + st]; sQ[tid] += sQ[tid + st]; }
        __syncthreads();
    }
    const float mu  = sS[0] * (1.0f / 768.0f);
    const float var = sQ[0] * (1.0f / 768.0f) - mu * mu;
    const float rs  = rsqrtf(var + 1e-5f);
    unsigned short* orow = out + (size_t)row * 768;
    orow[tid]       = f2bf((v0 - mu) * rs * g[tid]       + b[tid]);
    orow[tid + 256] = f2bf((v1 - mu) * rs * g[tid + 256] + b[tid + 256]);
    orow[tid + 512] = f2bf((v2 - mu) * rs * g[tid + 512] + b[tid + 512]);
}

// ---------------------------------------------------------------------------
// bf16 WMMA GEMM: C = A[M,K] @ Bt[N,K]^T.  Wave tile 32x64 (8 WMMA/k-step,
// B-fragment reuse x2), block = 8 waves = 64x256 tile.
// MODE 0: scatter into Q (scaled), K row-major, V transposed   (QKV gemm)
// MODE 1: outF = resid + acc + bias                            (merge / ffn2)
// MODE 2: outH = bf16(gelu(acc + bias))                        (ffn1)
// ---------------------------------------------------------------------------
template <int MODE>
__global__ __launch_bounds__(256) void tb_gemm_bf16(
    const unsigned short* __restrict__ A, const unsigned short* __restrict__ Bt,
    const float* __restrict__ bias, const float* __restrict__ resid,
    float* __restrict__ outF, unsigned short* __restrict__ outH,
    unsigned short* __restrict__ Qb, unsigned short* __restrict__ Kb,
    unsigned short* __restrict__ Vt, int M, int N, int K) {
    const int tid  = threadIdx.x;
    const int lane = tid & 31, wid = tid >> 5;
    const int ln   = lane & 15, hi = lane >> 4;
    const int rowBase = blockIdx.y * 64  + (wid & 1) * 32;
    const int colBase = blockIdx.x * 256 + (wid >> 1) * 64;

    v8f acc[2][4] = {{v8f{}, v8f{}, v8f{}, v8f{}}, {v8f{}, v8f{}, v8f{}, v8f{}}};
    const int aRow0 = rowBase + ln;
    const int aRow1 = rowBase + 16 + ln;

    for (int k0 = 0; k0 < K; k0 += 32) {
        v16bf fa0 = load_afrag(A, aRow0, K, k0, hi);
        v16bf fa1 = load_afrag(A, aRow1, K, k0, hi);
        if (k0 + 32 < K)   // pull next k-step's B column into cache
            __builtin_prefetch(Bt + (size_t)(colBase + ln) * K + k0 + 32, 0, 1);
#pragma unroll
        for (int j = 0; j < 4; ++j) {
            v16bf fb = load_bfrag(Bt, colBase + j * 16 + ln, K, k0, hi);
            acc[0][j] = wmma_bf16(fa0, fb, acc[0][j]);
            acc[1][j] = wmma_bf16(fa1, fb, acc[1][j]);
        }
    }

#pragma unroll
    for (int r = 0; r < 2; ++r) {
#pragma unroll
        for (int j = 0; j < 4; ++j) {
#pragma unroll
            for (int v = 0; v < 8; ++v) {
                const int   row = rowBase + r * 16 + v + 8 * hi;
                const int   col = colBase + j * 16 + ln;
                const float val = acc[r][j][v];
                if (MODE == 0) {
                    const int hd = col / 192, rr = col - hd * 192;
                    const int wh = rr >> 6, dd = rr & 63;
                    const int bb = row >> 11, ss = row & 2047;
                    const size_t bh = (size_t)(bb * 12 + hd);
                    if (wh == 0)      Qb[(bh * 2048 + ss) * 64 + dd] = f2bf(val * 0.125f);
                    else if (wh == 1) Kb[(bh * 2048 + ss) * 64 + dd] = f2bf(val);
                    else              Vt[(bh * 64 + dd) * 2048 + ss] = f2bf(val);
                } else if (MODE == 1) {
                    const size_t idx = (size_t)row * N + col;
                    outF[idx] = resid[idx] + val + bias[col];
                } else {
                    const float hh = val + bias[col];
                    const float ge = 0.5f * hh * (1.0f + erff(hh * 0.70710678f));
                    outH[(size_t)row * N + col] = f2bf(ge);
                }
            }
        }
    }
}

// ---------------------------------------------------------------------------
// Flash attention. Block = 4 waves = 64 query rows of one (b,h).
// 64 keys per iteration. K/V tiles staged in LDS via double-buffered
// global_load_async_to_lds_b128 (all 4 waves share the same K/V stream).
// Q pre-scaled by dh^-0.5; K [bh,N,dh] bf16; V transposed [bh,dh,N] bf16.
// ---------------------------------------------------------------------------
__global__ __launch_bounds__(128) void tb_flash_attn(
    const unsigned short* __restrict__ Qb, const unsigned short* __restrict__ Kb,
    const unsigned short* __restrict__ Vt, unsigned short* __restrict__ ctx) {
    __shared__ unsigned short Ks[2][64 * 64];   // [key][dh]     8KB x2
    __shared__ unsigned short Vs[2][64 * 64];   // [dh][key]     8KB x2
    __shared__ unsigned short Ps[4][16 * 64];   // per-wave P    8KB

    const int tid  = threadIdx.x;
    const int lane = tid & 31, wid = tid >> 5;
    const int ln   = lane & 15, hi = lane >> 4;
    const int bh   = blockIdx.x >> 5;
    const int q0   = (blockIdx.x & 31) * 64 + wid * 16;

    const size_t base = (size_t)bh * 2048 * 64;
    const unsigned short* Qp = Qb + base;
    const unsigned short* Kp = Kb + base;
    const unsigned short* Vp = Vt + base;
    unsigned short* myP = Ps[wid];

    // async-stage one 64-key K tile + V tile (512B/lane-instr, 8 instrs/wave)
    auto stage = [&](int buf, int kt) {
#pragma unroll
        for (int i = 0; i < 4; ++i) {
            const int c = i * 128 + tid;            // 16B chunk id, 0..511
            const int row = c >> 3, cof = (c & 7) * 8;
            async_cp16((unsigned)(size_t)&Ks[buf][row * 64 + cof],
                       (unsigned long long)(size_t)(Kp + (size_t)(kt + row) * 64 + cof));
            async_cp16((unsigned)(size_t)&Vs[buf][row * 64 + cof],
                       (unsigned long long)(size_t)(Vp + (size_t)row * 2048 + kt + cof));
        }
    };

    // Q A-fragments for dh chunks [0,32) and [32,64)
    const v16bf aq0 = load_afrag(Qp, q0 + ln, 64, 0,  hi);
    const v16bf aq1 = load_afrag(Qp, q0 + ln, 64, 32, hi);

    float mrow[8], lrow[8];
    v8f O[4] = {v8f{}, v8f{}, v8f{}, v8f{}};
#pragma unroll
    for (int v = 0; v < 8; ++v) { mrow[v] = -1e30f; lrow[v] = 0.0f; }

    stage(0, 0);
    for (int t = 0; t < 32; ++t) {
        const int buf = t & 1;
        const int kt  = t * 64;
        if (t < 31) {
            stage(buf ^ 1, kt + 64);
            asm volatile("s_wait_asynccnt 0x8" ::: "memory");   // tile t landed
        } else {
            asm volatile("s_wait_asynccnt 0x0" ::: "memory");
        }
        __syncthreads();                                        // visible to all waves

        // S = Q @ K^T for 64 keys: 4 column tiles x 2 dh chunks = 8 WMMA
        v8f S[4] = {v8f{}, v8f{}, v8f{}, v8f{}};
#pragma unroll
        for (int j = 0; j < 4; ++j) {
            S[j] = wmma_bf16(aq0, load_bfrag(Ks[buf], j * 16 + ln, 64, 0,  hi), S[j]);
            S[j] = wmma_bf16(aq1, load_bfrag(Ks[buf], j * 16 + ln, 64, 32, hi), S[j]);
        }

        // online softmax; reductions stay within each 16-lane half
#pragma unroll
        for (int v = 0; v < 8; ++v) {
            float s0 = S[0][v], s1 = S[1][v], s2 = S[2][v], s3 = S[3][v];
            float mx = fmaxf(fmaxf(s0, s1), fmaxf(s2, s3));
            mx = fmaxf(mx, __shfl_xor(mx, 1, 32));
            mx = fmaxf(mx, __shfl_xor(mx, 2, 32));
            mx = fmaxf(mx, __shfl_xor(mx, 4, 32));
            mx = fmaxf(mx, __shfl_xor(mx, 8, 32));
            const float mnew  = fmaxf(mrow[v], mx);
            const float alpha = __expf(mrow[v] - mnew);
            const float p0 = __expf(s0 - mnew), p1 = __expf(s1 - mnew);
            const float p2 = __expf(s2 - mnew), p3 = __expf(s3 - mnew);
            float rs = (p0 + p1) + (p2 + p3);
            rs += __shfl_xor(rs, 1, 32);
            rs += __shfl_xor(rs, 2, 32);
            rs += __shfl_xor(rs, 4, 32);
            rs += __shfl_xor(rs, 8, 32);
            lrow[v] = lrow[v] * alpha + rs;
            mrow[v] = mnew;
            O[0][v] *= alpha; O[1][v] *= alpha; O[2][v] *= alpha; O[3][v] *= alpha;
            const int pr = (v + 8 * hi) * 64;
            myP[pr + ln]      = f2bf(p0);
            myP[pr + 16 + ln] = f2bf(p1);
            myP[pr + 32 + ln] = f2bf(p2);
            myP[pr + 48 + ln] = f2bf(p3);
        }
        asm volatile("s_wait_dscnt 0x0" ::: "memory");  // cross-lane LDS hand-off

        // P A-fragments (16x64 -> two 16x32 chunks) from LDS
        BF16x16 ap0, ap1;
        const unsigned short* pr0 = myP + ln * 64 + 8 * hi;
        ap0.q[0] = *(const uint4*)(pr0);
        ap0.q[1] = *(const uint4*)(pr0 + 16);
        ap1.q[0] = *(const uint4*)(pr0 + 32);
        ap1.q[1] = *(const uint4*)(pr0 + 48);

        // O += P @ V : 4 dh tiles x 2 key chunks = 8 WMMA
#pragma unroll
        for (int j = 0; j < 4; ++j) {
            O[j] = wmma_bf16(ap0.v, load_bfrag(Vs[buf], j * 16 + ln, 64, 0,  hi), O[j]);
            O[j] = wmma_bf16(ap1.v, load_bfrag(Vs[buf], j * 16 + ln, 64, 32, hi), O[j]);
        }
        __syncthreads();   // all waves done with buf before it is re-staged
    }

    const int bb = bh / 12, hd = bh % 12;
#pragma unroll
    for (int v = 0; v < 8; ++v) {
        const float inv = 1.0f / lrow[v];
        const size_t rb = (size_t)(bb * 2048 + q0 + v + 8 * hi) * 768 + hd * 64;
        ctx[rb +  0 + ln] = f2bf(O[0][v] * inv);
        ctx[rb + 16 + ln] = f2bf(O[1][v] * inv);
        ctx[rb + 32 + ln] = f2bf(O[2][v] * inv);
        ctx[rb + 48 + ln] = f2bf(O[3][v] * inv);
    }
}

// ---------------------------------------------------------------------------
extern "C" void kernel_launch(void* const* d_in, const int* in_sizes, int n_in,
                              void* d_out, int out_size, void* d_ws, size_t ws_size,
                              hipStream_t stream) {
    const float* x       = (const float*)d_in[0];
    const float* w_qkv   = (const float*)d_in[1];
    const float* w_merge = (const float*)d_in[2];
    const float* b_merge = (const float*)d_in[3];
    const float* ln1_g   = (const float*)d_in[4];
    const float* ln1_b   = (const float*)d_in[5];
    const float* ln2_g   = (const float*)d_in[6];
    const float* ln2_b   = (const float*)d_in[7];
    const float* w1      = (const float*)d_in[8];
    const float* b1      = (const float*)d_in[9];
    const float* w2      = (const float*)d_in[10];
    const float* b2      = (const float*)d_in[11];
    float* out = (float*)d_out;

    const int M = 4 * 2048, D = 768, H3 = 2304, F = 3072;
    const size_t QKVE = (size_t)48 * 2048 * 64;

    char* base = (char*)d_ws;
    size_t off = 0;
    auto get = [&](size_t elems, size_t esz) -> void* {
        void* r = base + off;
        off = (off + elems * esz + 255) & ~(size_t)255;
        return r;
    };
    unsigned short* xn1   = (unsigned short*)get((size_t)M * D, 2);
    unsigned short* wqkvT = (unsigned short*)get((size_t)H3 * D, 2);
    unsigned short* wmT   = (unsigned short*)get((size_t)D * D, 2);
    unsigned short* w1T   = (unsigned short*)get((size_t)F * D, 2);
    unsigned short* w2T   = (unsigned short*)get((size_t)D * F, 2);
    unsigned short* Qb    = (unsigned short*)get(QKVE, 2);
    unsigned short* Kb    = (unsigned short*)get(QKVE, 2);
    unsigned short* Vt    = (unsigned short*)get(QKVE, 2);
    unsigned short* ctx   = (unsigned short*)get((size_t)M * D, 2);
    float*          x2    = (float*)get((size_t)M * D, 4);
    unsigned short* xn2   = (unsigned short*)get((size_t)M * D, 2);
    unsigned short* hdn   = (unsigned short*)get((size_t)M * F, 2);

    auto blks = [](long n) { return (unsigned)((n + 255) / 256); };

    // weight prep (bf16, transposed to [N][K])
    tb_transpose_cast<<<blks((long)D * H3), 256, 0, stream>>>(w_qkv, wqkvT, D, H3);
    tb_transpose_cast<<<blks((long)D * D),  256, 0, stream>>>(w_merge, wmT, D, D);
    tb_transpose_cast<<<blks((long)D * F),  256, 0, stream>>>(w1, w1T, D, F);
    tb_transpose_cast<<<blks((long)F * D),  256, 0, stream>>>(w2, w2T, F, D);

    // LN1
    tb_layernorm<<<M, 256, 0, stream>>>(x, ln1_g, ln1_b, xn1);

    // QKV gemm + scatter (Q scaled, V transposed)
    tb_gemm_bf16<0><<<dim3(H3 / 256, M / 64), 256, 0, stream>>>(
        xn1, wqkvT, nullptr, nullptr, nullptr, nullptr, Qb, Kb, Vt, M, H3, D);

    // flash attention
    tb_flash_attn<<<48 * 32, 128, 0, stream>>>(Qb, Kb, Vt, ctx);

    // merge + residual
    tb_gemm_bf16<1><<<dim3(D / 256, M / 64), 256, 0, stream>>>(
        ctx, wmT, b_merge, x, x2, nullptr, nullptr, nullptr, nullptr, M, D, D);

    // LN2
    tb_layernorm<<<M, 256, 0, stream>>>(x2, ln2_g, ln2_b, xn2);

    // FFN1 + GELU
    tb_gemm_bf16<2><<<dim3(F / 256, M / 64), 256, 0, stream>>>(
        xn2, w1T, b1, nullptr, nullptr, hdn, nullptr, nullptr, nullptr, M, F, D);

    // FFN2 + residual -> out
    tb_gemm_bf16<1><<<dim3(D / 256, M / 64), 256, 0, stream>>>(
        hdn, w2T, b2, x2, out, nullptr, nullptr, nullptr, nullptr, M, D, F);

    (void)in_sizes; (void)n_in; (void)out_size; (void)ws_size;
}
